// RadialBasisFunctionSampler3D_7421703487943
// MI455X (gfx1250) — matrix operand order repaired and checked by reading, not compiled, not run on hardware
//
#include <hip/hip_runtime.h>

typedef __attribute__((ext_vector_type(16))) __bf16        v16bf;
typedef __attribute__((ext_vector_type(8)))  float         v8f;
typedef __attribute__((ext_vector_type(8)))  unsigned int  v8u;

#define NPTS 50000
#define NCH  512
#define NG   512
#define LOG2E 1.4426950408889634f

__device__ __forceinline__ float fast_exp2(float x) {
#if __has_builtin(__builtin_amdgcn_exp2f)
    return __builtin_amdgcn_exp2f(x);     // v_exp_f32
#else
    return exp2f(x);
#endif
}

// One-time: convert latents fp32 -> bf16 into d_ws, pre-swizzled into WMMA
// B-fragment order: ws[b][kc][nt][lane][j] with
//   k = kc*32 + (lane>>4)*16 + j ,  n = nt*16 + (lane&15)
// so each lane's 16 bf16 (32B) fragment is one contiguous coalesced load.
__global__ __launch_bounds__(256) void rbf_swizzleB_kernel(
    const float* __restrict__ lat, v16bf* __restrict__ wsB)
{
    int t    = blockIdx.x * 256 + threadIdx.x;   // 65536 threads total
    int lane = t & 31;
    int nt   = (t >> 5) & 31;
    int kc   = (t >> 10) & 15;
    int b    = (t >> 14) & 3;
    int n     = nt * 16 + (lane & 15);
    int kbase = kc * 32 + ((lane >> 4) << 4);
    const float* src = lat + (size_t)b * (NG * NCH) + (size_t)kbase * NCH + n;
    v16bf v;
#pragma unroll
    for (int j = 0; j < 16; ++j) {
        v[j] = (__bf16)src[j * NCH];     // fptrunc f32 -> bf16 (RNE)
    }
    wsB[t] = v;
}

// Main kernel: 8 waves/block; wave w -> 16 points, 128 channels (8 WMMA tiles).
__global__ __launch_bounds__(256) void rbf_wmma_kernel(
    const float* __restrict__ points,
    const v16bf* __restrict__ wsB,
    const float* __restrict__ eps,
    float* __restrict__ out)
{
    __shared__ float4 tab[NG];   // log2e * (-2*e2*gx, -2*e2*gy, -2*e2*gz, e2*g2)
    __shared__ float  e2t[NG];   // log2e * e2

    const int b       = blockIdx.z;
    const int n0      = blockIdx.y * 128;
    const int m_block = blockIdx.x * 128;

    // Per-gridpoint RBF constants, pre-scaled by log2(e) so A = exp2(-t).
    for (int k = threadIdx.x; k < NG; k += 256) {
        float e  = eps[k];
        float ee = e * e * LOG2E;
        float gx = -1.0f + (float)(k >> 6)       * (2.0f / 7.0f);
        float gy = -1.0f + (float)((k >> 3) & 7) * (2.0f / 7.0f);
        float gz = -1.0f + (float)(k & 7)        * (2.0f / 7.0f);
        float g2 = gx * gx + gy * gy + gz * gz;
        tab[k] = make_float4(-2.0f * ee * gx, -2.0f * ee * gy, -2.0f * ee * gz, ee * g2);
        e2t[k] = ee;
    }
    __syncthreads();

    const int wave = threadIdx.x >> 5;
    const int lane = threadIdx.x & 31;
    const int hi   = lane >> 4;       // half-wave select
    const int r    = lane & 15;       // row within tile

    const int m0  = m_block + wave * 16;
    int pm        = m0 + r;
    int pmc       = pm < NPTS ? pm : (NPTS - 1);
    const float* pp = points + (size_t)(b * NPTS + pmc) * 3;
    const float px = pp[0], py = pp[1], pz = pp[2];
    const float p2 = px * px + py * py + pz * pz;

    v8f acc[8] = {};

    const int ntbase = blockIdx.y * 8;         // first 16-col tile index
    const v16bf* bbase = wsB + (size_t)lane;

    for (int kc = 0; kc < 16; ++kc) {
        // ---- Generate A fragment (RBF exp2 values) directly in register form, k = 16*hi + j (as B) ----
        v16bf afrag;
#pragma unroll
        for (int j = 0; j < 16; ++j) {
            int kl = j + (hi << 4);
            int k  = kc * 32 + kl;
            float4 tv = tab[k];
            float  t  = fmaf(e2t[k], p2, tv.w);   // log2e*e2*(p2+g2)
            t = fmaf(tv.x, px, t);
            t = fmaf(tv.y, py, t);
            t = fmaf(tv.z, pz, t);                // log2e*e2*d2
            t = fmaxf(t, 0.0f);
            afrag[j] = (__bf16)fast_exp2(-t);     // fptrunc -> packed cvt
        }

        // ---- 8 WMMA tiles across 128 channels, B fragments from d_ws ----
#pragma unroll
        for (int tt = 0; tt < 8; ++tt) {
            int nt = ntbase + tt;
            v16bf bfrag = bbase[(size_t)(((b * 16 + kc) * 32 + nt)) * 32];
            acc[tt] = __builtin_amdgcn_wmma_f32_16x16x32_bf16(
                false, afrag, false, bfrag, (short)0, acc[tt], false, false);
        }
    }

    // ---- Store D tiles: lane r holds column n0+tt*16+r; VGPR v -> row m0+v+hi*8
    float* obase = out + (((size_t)(b * NPTS + m0 + hi * 8)) << 9) + n0 + r;
    if (m0 + 16 <= NPTS) {
        // Fast path: whole tile in range, no per-element guards.
#pragma unroll
        for (int tt = 0; tt < 8; ++tt) {
#pragma unroll
            for (int v = 0; v < 8; ++v) {
                obase[((size_t)v << 9) + tt * 16] = acc[tt][v];
            }
        }
    } else {
#pragma unroll
        for (int tt = 0; tt < 8; ++tt) {
#pragma unroll
            for (int v = 0; v < 8; ++v) {
                int m = m0 + v + hi * 8;
                if (m < NPTS)
                    obase[((size_t)v << 9) + tt * 16] = acc[tt][v];
            }
        }
    }
}

extern "C" void kernel_launch(void* const* d_in, const int* in_sizes, int n_in,
                              void* d_out, int out_size, void* d_ws, size_t ws_size,
                              hipStream_t stream) {
    const float* points = (const float*)d_in[0];   // [4, 50000, 3]
    const float* lat    = (const float*)d_in[1];   // [4, 512, 8, 8, 8] -> [4,512,512]
    const float* eps    = (const float*)d_in[2];   // [512]
    float* out          = (float*)d_out;           // [4, 50000, 512]
    v16bf* wsB          = (v16bf*)d_ws;            // 2 MB bf16 swizzled B

    // 1) bf16 convert + fragment-swizzle latents into workspace
    rbf_swizzleB_kernel<<<256, 256, 0, stream>>>(lat, wsB);

    // 2) RBF + batched GEMM via WMMA
    dim3 grid((NPTS + 127) / 128, NCH / 128, 4);
    rbf_wmma_kernel<<<grid, 256, 0, stream>>>(points, wsB, eps, out);
}
